// PainnModel_41291815584026
// MI455X (gfx1250) — matrix operand order repaired and checked by reading, not compiled, not run on hardware
//
#include <hip/hip_runtime.h>
#include <hip/hip_bf16.h>
#include <math.h>

// ---------------------------------------------------------------------------
// PaiNN forward for MI455X (gfx1250, wave32).
// Dense per-atom GEMMs: v_wmma_f32_16x16x32_f16, one 16x64 C strip per wave.
// Weights pre-transposed AND pre-converted to f16 (N x K) so B fragments are
// direct v16h loads (2x global_load_b128, no cvt). Inner loop is manually
// software-pipelined: next k-step's A/B loads are issued into fresh registers
// before the current 4 WMMAs, so loads batch under a single s_wait and overlap
// with matrix math.
// Edge message+scatter: fused kernel with global fp32 atomics.
// Workspace: ~50 MB of d_ws.
// ---------------------------------------------------------------------------

typedef __attribute__((ext_vector_type(16))) _Float16 v16h;
typedef __attribute__((ext_vector_type(8)))  float    v8f;

#define N_ATOMS 5000
#define N_EDGES 100000
#define FDIM    128
#define BDIM    20
#define LAYERS  3
#define CUTOFF  5.0f
#define PI_F    3.14159265358979f

__device__ __forceinline__ float silu_f(float x) { return x / (1.0f + __expf(-x)); }

__device__ __forceinline__ v16h cvt16(float4 x0, float4 x1, float4 x2, float4 x3) {
    v16h r;
    r[0]  = (_Float16)x0.x; r[1]  = (_Float16)x0.y; r[2]  = (_Float16)x0.z; r[3]  = (_Float16)x0.w;
    r[4]  = (_Float16)x1.x; r[5]  = (_Float16)x1.y; r[6]  = (_Float16)x1.z; r[7]  = (_Float16)x1.w;
    r[8]  = (_Float16)x2.x; r[9]  = (_Float16)x2.y; r[10] = (_Float16)x2.z; r[11] = (_Float16)x2.w;
    r[12] = (_Float16)x3.x; r[13] = (_Float16)x3.y; r[14] = (_Float16)x3.z; r[15] = (_Float16)x3.w;
    return r;
}

// ---------------------------------------------------------------------------
// C[M x N] = act( A[M x K(lda)] @ W[K x N] + bias ), W given as f16, N x K.
// A fragment (16x32 f16): lane&15 = row; elems 0..15 = K k0+16h..+15
//                         (one 16-float run).
// B fragment (32x16 f16): lane&15 = col; elems 0..15 = K k0+16h..+15 (one run).
// One wave computes a 16x64 strip: 4 accumulators, 4 WMMAs per 32-wide k-step.
// Requires: K % 32 == 0, N % 64 == 0.
// ---------------------------------------------------------------------------
template <int ACT>
__global__ __launch_bounds__(128) void wmma_gemm_kernel(
    const float* __restrict__ A, int lda,
    const _Float16* __restrict__ Wt,
    const float* __restrict__ bias,
    float* __restrict__ C,
    int M, int N, int K) {
    const int lane   = threadIdx.x & 31;
    const int wave   = blockIdx.x * (blockDim.x >> 5) + (threadIdx.x >> 5);
    const int tilesN = N >> 6;
    const int tilesM = (M + 15) >> 4;
    if (wave >= tilesM * tilesN) return;  // whole-wave uniform exit

    const int m0   = (wave / tilesN) << 4;
    const int n0   = (wave % tilesN) << 6;
    const int half = lane >> 4;
    const int r15  = lane & 15;

    int arow = m0 + r15;
    if (arow >= M) arow = M - 1;  // clamp: padded rows never stored

    const float*    ap = A  + (size_t)arow * lda + 16 * half;
    const _Float16* wp = Wt + (size_t)(n0 + r15) * K + 16 * half;
    const size_t    cs = (size_t)16 * K;  // 16 weight columns

    // Prologue: k-step 0 loads.
    float4 ax0 = *(const float4*)(ap + 0);
    float4 ax1 = *(const float4*)(ap + 4);
    float4 ax2 = *(const float4*)(ap + 8);
    float4 ax3 = *(const float4*)(ap + 12);
    v16h   b0  = *(const v16h*)(wp);
    v16h   b1  = *(const v16h*)(wp + cs);
    v16h   b2  = *(const v16h*)(wp + 2 * cs);
    v16h   b3  = *(const v16h*)(wp + 3 * cs);

    v8f acc0 = {}, acc1 = {}, acc2 = {}, acc3 = {};
    for (int k0 = 0;;) {
        const bool more = (k0 + 32 < K);  // wave-uniform
        float4 nx0, nx1, nx2, nx3;
        v16h   nb0, nb1, nb2, nb3;
        if (more) {  // issue next k-step's loads before this step's WMMAs
            nx0 = *(const float4*)(ap + 32);
            nx1 = *(const float4*)(ap + 36);
            nx2 = *(const float4*)(ap + 40);
            nx3 = *(const float4*)(ap + 44);
            nb0 = *(const v16h*)(wp + 32);
            nb1 = *(const v16h*)(wp + 32 + cs);
            nb2 = *(const v16h*)(wp + 32 + 2 * cs);
            nb3 = *(const v16h*)(wp + 32 + 3 * cs);
        }
        const v16h a = cvt16(ax0, ax1, ax2, ax3);
        acc0 = __builtin_amdgcn_wmma_f32_16x16x32_f16(false, a, false, b0, (short)0, acc0, false, false);
        acc1 = __builtin_amdgcn_wmma_f32_16x16x32_f16(false, a, false, b1, (short)0, acc1, false, false);
        acc2 = __builtin_amdgcn_wmma_f32_16x16x32_f16(false, a, false, b2, (short)0, acc2, false, false);
        acc3 = __builtin_amdgcn_wmma_f32_16x16x32_f16(false, a, false, b3, (short)0, acc3, false, false);
        if (!more) break;
        ax0 = nx0; ax1 = nx1; ax2 = nx2; ax3 = nx3;
        b0 = nb0; b1 = nb1; b2 = nb2; b3 = nb3;
        ap += 32; wp += 32; k0 += 32;
    }

#pragma unroll
    for (int r = 0; r < 8; ++r) {
        const int row = m0 + half * 8 + r;
        if (row < M) {
            float* crow = C + (size_t)row * N + n0 + r15;
            float v0 = acc0[r], v1 = acc1[r], v2 = acc2[r], v3 = acc3[r];
            if (bias) {
                v0 += bias[n0 + r15];
                v1 += bias[n0 + 16 + r15];
                v2 += bias[n0 + 32 + r15];
                v3 += bias[n0 + 48 + r15];
            }
            if (ACT) { v0 = silu_f(v0); v1 = silu_f(v1); v2 = silu_f(v2); v3 = silu_f(v3); }
            crow[0] = v0; crow[16] = v1; crow[32] = v2; crow[48] = v3;
        }
    }
}

// out[N x K] (f16) = transpose(in[K x N]) (f32)
__global__ void transpose_cvt_kernel(const float* __restrict__ in,
                                     _Float16* __restrict__ out, int K, int N) {
    int idx = blockIdx.x * blockDim.x + threadIdx.x;
    if (idx >= K * N) return;
    int k = idx / N, n = idx - k * N;
    out[(size_t)n * K + k] = (_Float16)in[idx];
}

// ---------------------------------------------------------------------------
// Elementwise / init / edge kernels
// ---------------------------------------------------------------------------
__global__ void init_sfeat_kernel(const int* __restrict__ Z,
                                  const float* __restrict__ emb,
                                  float* __restrict__ s_feat) {
    int idx = blockIdx.x * blockDim.x + threadIdx.x;
    if (idx >= N_ATOMS * FDIM) return;
    int n = idx >> 7, f = idx & 127;
    s_feat[idx] = emb[Z[n] * FDIM + f];
}

__global__ void zero_kernel(float* __restrict__ p, int n) {
    int idx = blockIdx.x * blockDim.x + threadIdx.x;
    if (idx < n) p[idx] = 0.0f;
}

// One block (384 threads) per edge: filter GEMV + gather(phi, v_prev) + scatter.
__global__ __launch_bounds__(384) void edge_message_kernel(
    const int* __restrict__ edge_idx,
    const float* __restrict__ edge_diff,
    const float* __restrict__ edge_dist,
    const float* __restrict__ filt_w,  // 20 x 384
    const float* __restrict__ filt_b,  // 384
    const float* __restrict__ phi,     // N_ATOMS x 384
    const float* __restrict__ v_prev,  // N_ATOMS x 3 x 128
    float* __restrict__ s_feat,        // atomic +=
    float* __restrict__ v_feat) {      // atomic +=
    __shared__ float rbf_s[BDIM];
    __shared__ float gsv_s[FDIM];
    __shared__ float gev_s[FDIM];
    __shared__ float env_s;
    __shared__ float unit_s[3];

    const int e = blockIdx.x;
    const int t = threadIdx.x;  // 0..383
    const float d = edge_dist[e];

    if (t < BDIM) rbf_s[t] = __sinf(d * (float)(t + 1) * (PI_F / CUTOFF)) / d;
    if (t == BDIM) {
        float fc = 0.5f * (__cosf(PI_F * d / CUTOFF) + 1.0f);
        env_s = (d < CUTOFF) ? fc : 0.0f;
    }
    if (t >= BDIM + 1 && t < BDIM + 4) {
        int i = t - (BDIM + 1);
        unit_s[i] = edge_diff[e * 3 + i] / d;
    }
    __syncthreads();

    const int dst = edge_idx[e * 2 + 0];
    const int src = edge_idx[e * 2 + 1];

    float f = filt_b[t];
#pragma unroll
    for (int b = 0; b < BDIM; ++b) f += rbf_s[b] * filt_w[b * 384 + t];
    const float fo = f * env_s * phi[(size_t)src * 384 + t];

    if (t < FDIM)          gsv_s[t] = fo;
    else if (t < 2 * FDIM) gev_s[t - FDIM] = fo;
    else                   atomicAdd(&s_feat[(size_t)dst * FDIM + (t - 2 * FDIM)], fo);
    __syncthreads();

    const int i = t >> 7, ff = t & 127;  // (i, f) over 3 x 128
    const float mv = v_prev[((size_t)src * 3 + i) * FDIM + ff] * gsv_s[ff]
                   + gev_s[ff] * unit_s[i];
    atomicAdd(&v_feat[((size_t)dst * 3 + i) * FDIM + ff], mv);
}

// cat[n][0:128) = s_feat ; cat[n][128:256) = ||Vv[n][:, g]||
__global__ void vn_cat_kernel(const float* __restrict__ Vv,
                              const float* __restrict__ s_feat,
                              float* __restrict__ cat) {
    int idx = blockIdx.x * blockDim.x + threadIdx.x;
    if (idx >= N_ATOMS * FDIM) return;
    int n = idx >> 7, g = idx & 127;
    float a0 = Vv[((size_t)n * 3 + 0) * FDIM + g];
    float a1 = Vv[((size_t)n * 3 + 1) * FDIM + g];
    float a2 = Vv[((size_t)n * 3 + 2) * FDIM + g];
    cat[(size_t)n * 256 + g]        = s_feat[idx];
    cat[(size_t)n * 256 + FDIM + g] = sqrtf(a0 * a0 + a1 * a1 + a2 * a2);
}

// v += avv*Uv ; s += asv*sum_i(Uv*Vv) + ass
__global__ void update_kernel(const float* __restrict__ a_out,  // N_ATOMS x 384
                              const float* __restrict__ Uv,
                              const float* __restrict__ Vv,
                              float* __restrict__ s_feat,
                              float* __restrict__ v_feat) {
    int idx = blockIdx.x * blockDim.x + threadIdx.x;
    if (idx >= N_ATOMS * FDIM) return;
    int n = idx >> 7, f = idx & 127;
    const float avv = a_out[(size_t)n * 384 + f];
    const float asv = a_out[(size_t)n * 384 + 128 + f];
    const float ass = a_out[(size_t)n * 384 + 256 + f];
    float dot = 0.0f;
#pragma unroll
    for (int i = 0; i < 3; ++i) {
        const size_t off = ((size_t)n * 3 + i) * FDIM + f;
        const float u = Uv[off];
        dot += u * Vv[off];
        v_feat[off] += avv * u;
    }
    s_feat[idx] += asv * dot + ass;
}

// out[n] = hid[n,:] . ro_w2 + ro_b2
__global__ void readout_kernel(const float* __restrict__ hid,   // N_ATOMS x 64
                               const float* __restrict__ ro_w2, // 64
                               const float* __restrict__ ro_b2, // 1
                               float* __restrict__ out) {
    int n = blockIdx.x * blockDim.x + threadIdx.x;
    if (n >= N_ATOMS) return;
    float s = ro_b2[0];
#pragma unroll
    for (int g = 0; g < 64; ++g) s += hid[(size_t)n * 64 + g] * ro_w2[g];
    out[n] = s;
}

// ---------------------------------------------------------------------------
// Host orchestration
// ---------------------------------------------------------------------------
static inline void launch_gemm(int act, const float* A, int lda, const _Float16* Wt,
                               const float* bias, float* C, int M, int N, int K,
                               hipStream_t s) {
    const int tiles  = ((M + 15) / 16) * (N / 64);
    const int blocks = (tiles + 3) / 4;  // 4 waves (128 threads) per block
    if (act)
        wmma_gemm_kernel<1><<<blocks, 128, 0, s>>>(A, lda, Wt, bias, C, M, N, K);
    else
        wmma_gemm_kernel<0><<<blocks, 128, 0, s>>>(A, lda, Wt, bias, C, M, N, K);
}

static inline void launch_transpose(const float* in, _Float16* out, int K, int N,
                                    hipStream_t s) {
    const int n = K * N;
    transpose_cvt_kernel<<<(n + 255) / 256, 256, 0, s>>>(in, out, K, N);
}

extern "C" void kernel_launch(void* const* d_in, const int* in_sizes, int n_in,
                              void* d_out, int out_size, void* d_ws, size_t ws_size,
                              hipStream_t stream) {
    (void)in_sizes; (void)n_in; (void)out_size; (void)ws_size;

    const int*   Z         = (const int*)d_in[0];
    const int*   edge_idx  = (const int*)d_in[1];
    const float* edge_diff = (const float*)d_in[2];
    const float* edge_dist = (const float*)d_in[3];
    const float* emb       = (const float*)d_in[4];
    const float* msg_w1    = (const float*)d_in[5];   // L x 128 x 128
    const float* msg_b1    = (const float*)d_in[6];   // L x 128
    const float* msg_w2    = (const float*)d_in[7];   // L x 128 x 384
    const float* msg_b2    = (const float*)d_in[8];   // L x 384
    const float* filt_w    = (const float*)d_in[9];   // L x 20 x 384
    const float* filt_b    = (const float*)d_in[10];  // L x 384
    const float* upd_U     = (const float*)d_in[11];  // L x 128 x 128
    const float* upd_V     = (const float*)d_in[12];  // L x 128 x 128
    const float* upd_w1    = (const float*)d_in[13];  // L x 256 x 128
    const float* upd_b1    = (const float*)d_in[14];  // L x 128
    const float* upd_w2    = (const float*)d_in[15];  // L x 128 x 384
    const float* upd_b2    = (const float*)d_in[16];  // L x 384
    const float* ro_w1     = (const float*)d_in[17];  // 128 x 64
    const float* ro_b1     = (const float*)d_in[18];  // 64
    const float* ro_w2     = (const float*)d_in[19];  // 64 x 1
    const float* ro_b2     = (const float*)d_in[20];  // 1
    float*       out       = (float*)d_out;

    // Workspace carve-up. f32 activations first (every size is a multiple of
    // 64 floats -> 256B alignment preserved), then f16 transposed weights
    // (every size a multiple of 16 halves -> 32B alignment preserved).
    float* ws = (float*)d_ws;
    size_t o = 0;
    float* s_feat = ws + o; o += (size_t)N_ATOMS * FDIM;
    float* v_feat = ws + o; o += (size_t)N_ATOMS * 3 * FDIM;
    float* v_prev = ws + o; o += (size_t)N_ATOMS * 3 * FDIM;
    float* phi    = ws + o; o += (size_t)N_ATOMS * 384;   // reused as a_out
    float* hid    = ws + o; o += (size_t)N_ATOMS * FDIM;
    float* Uv     = ws + o; o += (size_t)N_ATOMS * 3 * FDIM;
    float* Vv     = ws + o; o += (size_t)N_ATOMS * 3 * FDIM;
    float* cat    = ws + o; o += (size_t)N_ATOMS * 256;

    _Float16* hws = (_Float16*)(ws + o);
    size_t h = 0;
    _Float16* mw1t  = hws + h; h += (size_t)LAYERS * FDIM * FDIM;  // 128x128 -> 128x128
    _Float16* mw2t  = hws + h; h += (size_t)LAYERS * FDIM * 384;   // 128x384 -> 384x128
    _Float16* uUt   = hws + h; h += (size_t)LAYERS * FDIM * FDIM;
    _Float16* uVt   = hws + h; h += (size_t)LAYERS * FDIM * FDIM;
    _Float16* uw1t  = hws + h; h += (size_t)LAYERS * 256 * FDIM;   // 256x128 -> 128x256
    _Float16* uw2t  = hws + h; h += (size_t)LAYERS * FDIM * 384;
    _Float16* row1t = hws + h; h += (size_t)FDIM * 64;             // 128x64 -> 64x128

    const int NF = N_ATOMS * FDIM;
    const int NV = N_ATOMS * 3 * FDIM;

    // Pre-transpose + f16-convert all weights (small, once per call).
    for (int l = 0; l < LAYERS; ++l) {
        launch_transpose(msg_w1 + (size_t)l * FDIM * FDIM, mw1t + (size_t)l * FDIM * FDIM, FDIM, FDIM, stream);
        launch_transpose(msg_w2 + (size_t)l * FDIM * 384,  mw2t + (size_t)l * FDIM * 384,  FDIM, 384,  stream);
        launch_transpose(upd_U  + (size_t)l * FDIM * FDIM, uUt  + (size_t)l * FDIM * FDIM, FDIM, FDIM, stream);
        launch_transpose(upd_V  + (size_t)l * FDIM * FDIM, uVt  + (size_t)l * FDIM * FDIM, FDIM, FDIM, stream);
        launch_transpose(upd_w1 + (size_t)l * 256 * FDIM,  uw1t + (size_t)l * 256 * FDIM,  256,  FDIM, stream);
        launch_transpose(upd_w2 + (size_t)l * FDIM * 384,  uw2t + (size_t)l * FDIM * 384,  FDIM, 384,  stream);
    }
    launch_transpose(ro_w1, row1t, FDIM, 64, stream);

    init_sfeat_kernel<<<(NF + 255) / 256, 256, 0, stream>>>(Z, emb, s_feat);
    zero_kernel<<<(NV + 255) / 256, 256, 0, stream>>>(v_feat, NV);

    for (int l = 0; l < LAYERS; ++l) {
        const float* mb1 = msg_b1 + (size_t)l * FDIM;
        const float* mb2 = msg_b2 + (size_t)l * 384;
        const float* fw  = filt_w + (size_t)l * BDIM * 384;
        const float* fb  = filt_b + (size_t)l * 384;
        const float* ub1 = upd_b1 + (size_t)l * FDIM;
        const float* ub2 = upd_b2 + (size_t)l * 384;

        // phi = silu(s @ mw1 + mb1) @ mw2 + mb2
        launch_gemm(1, s_feat, FDIM, mw1t + (size_t)l * FDIM * FDIM, mb1, hid, N_ATOMS, FDIM, FDIM, stream);
        launch_gemm(0, hid,   FDIM, mw2t + (size_t)l * FDIM * 384,  mb2, phi, N_ATOMS, 384,  FDIM, stream);

        // snapshot v for the gather side of the edge scatter
        hipMemcpyAsync(v_prev, v_feat, (size_t)NV * sizeof(float),
                       hipMemcpyDeviceToDevice, stream);

        edge_message_kernel<<<N_EDGES, 384, 0, stream>>>(
            edge_idx, edge_diff, edge_dist, fw, fb, phi, v_prev, s_feat, v_feat);

        // Uv / Vv : (N*3) x 128 @ 128 x 128
        launch_gemm(0, v_feat, FDIM, uUt + (size_t)l * FDIM * FDIM, nullptr, Uv, N_ATOMS * 3, FDIM, FDIM, stream);
        launch_gemm(0, v_feat, FDIM, uVt + (size_t)l * FDIM * FDIM, nullptr, Vv, N_ATOMS * 3, FDIM, FDIM, stream);

        vn_cat_kernel<<<(NF + 255) / 256, 256, 0, stream>>>(Vv, s_feat, cat);

        // a = silu(cat @ uw1 + ub1) @ uw2 + ub2   (phi buffer reused as a_out)
        launch_gemm(1, cat, 256, uw1t + (size_t)l * 256 * FDIM, ub1, hid, N_ATOMS, FDIM, 256, stream);
        launch_gemm(0, hid, FDIM, uw2t + (size_t)l * FDIM * 384, ub2, phi, N_ATOMS, 384, FDIM, stream);

        update_kernel<<<(NF + 255) / 256, 256, 0, stream>>>(phi, Uv, Vv, s_feat, v_feat);
    }

    // readout: hid = silu(s @ ro_w1 + ro_b1) ; out = hid @ ro_w2 + ro_b2
    launch_gemm(1, s_feat, FDIM, row1t, ro_b1, hid, N_ATOMS, 64, FDIM, stream);
    readout_kernel<<<(N_ATOMS + 255) / 256, 256, 0, stream>>>(hid, ro_w2, ro_b2, out);
}
